// vanilla_lstm_67061619359898
// MI455X (gfx1250) — matrix operand order repaired and checked by reading, not compiled, not run on hardware
//
#include <hip/hip_runtime.h>

typedef __attribute__((ext_vector_type(16))) _Float16 v16h;
typedef __attribute__((ext_vector_type(8)))  float    v8f;
typedef __attribute__((ext_vector_type(4)))  unsigned int u32x4;
typedef __attribute__((ext_vector_type(8)))  int          i32x8;
typedef __attribute__((ext_vector_type(4)))  int          i32x4;

#define B_   256
#define V_   32
#define T_   48
#define P_   12
#define F_   4
#define H_   128
#define N_   (B_*V_)          // 8192 recurrent lanes
#define G_   (4*H_)           // 512 gate columns
#define ROWS 32               // rows per workgroup
#define KEXT 160              // K = 128 (h) + 4 (x) + 28 zero-pad -> 5 WMMA K-chunks
#define PITCH 168             // padded halves per row: 336B = 21*16B, bank-spread

// ---- LDS layout (bytes), all offsets 16B aligned ----
static constexpr size_t OFF_WHH  = 0;
static constexpr size_t SZ_WHH   = (size_t)G_ * PITCH * 2;       // 172032  [Whh|Wih|0] f16
static constexpr size_t OFF_WOUT = OFF_WHH + SZ_WHH;
static constexpr size_t SZ_WOUT  = (size_t)16 * PITCH * 2;       // 5376 (rows 4..15 zero)
static constexpr size_t OFF_H    = OFF_WOUT + SZ_WOUT;
static constexpr size_t SZ_H     = (size_t)ROWS * PITCH * 2;     // 10752  [h|x|0] f16
static constexpr size_t OFF_SEQ  = OFF_H + SZ_H;
static constexpr size_t SZ_SEQ   = (size_t)ROWS * T_ * F_ * 4;   // 24576  f32 sequence tile
static constexpr size_t OFF_BIAS = OFF_SEQ + SZ_SEQ;
static constexpr size_t SZ_BIAS  = (size_t)G_ * 4;               // 2048   b_ih + b_hh
static constexpr size_t OFF_BOUT = OFF_BIAS + SZ_BIAS;
static constexpr size_t SZ_BOUT  = 16;
static constexpr size_t SMEM_TOTAL = OFF_BOUT + SZ_BOUT;         // ~210 KB < 320 KB/WGP

// gfx1250 has native v_tanh_f32; sigmoid = 0.5*tanh(0.5x)+0.5 (1 TRANS + 1 FMA)
__device__ __forceinline__ float tanhfast(float x) {
#if __has_builtin(__builtin_amdgcn_tanhf)
    return __builtin_amdgcn_tanhf(x);
#else
    float e = __expf(2.0f * x);
    return (e - 1.0f) * __builtin_amdgcn_rcpf(e + 1.0f);
#endif
}
__device__ __forceinline__ float sigf(float x) {
    return __builtin_fmaf(0.5f, tanhfast(0.5f * x), 0.5f);
}

// A-fragment (16x32 f16): lane<16 row=ln holds halves [k0..k0+15];
// lane>=16 same row, k0 shifted by 16 (caller passes k0 = kc*32 + half*16).
__device__ __forceinline__ v16h load_a16(const _Float16* rowp, int k0) {
    v16h r;
    ((float4*)&r)[0] = *(const float4*)(rowp + k0);
    ((float4*)&r)[1] = *(const float4*)(rowp + k0 + 8);
    return r;
}
// B-fragment (32x16 f16): lane holds col=ln, 16 consecutive K halves.
__device__ __forceinline__ v16h load_b16(const _Float16* p) {
    v16h r;
    ((float4*)&r)[0] = ((const float4*)p)[0];
    ((float4*)&r)[1] = ((const float4*)p)[1];
    return r;
}

// Tensor Data Mover: 2D f32 tile [32 rows x 192 cols], stride 192, global -> LDS.
__device__ __forceinline__ void tdm_load_seq_tile(const float* gaddr, unsigned lds_off) {
    unsigned long long ga = (unsigned long long)(uintptr_t)gaddr;
    u32x4 g0;
    g0[0] = 1u;                                   // count=1, is_restore=0, no gather
    g0[1] = lds_off;                              // lds_addr (bytes)
    g0[2] = (unsigned)(ga & 0xffffffffu);         // global_addr[31:0]
    g0[3] = (unsigned)((ga >> 32) & 0x01ffffffu)  // global_addr[56:32]
          | 0x80000000u;                          // type=2 ("image")
    i32x8 g1;
    g1[0] = 0x00020000;                           // data_size=2 (4B), mask=0, no pad
    g1[1] = (int)((unsigned)(T_ * F_) << 16);     // tensor_dim0 = 192 (lo16)
    g1[2] = (int)((unsigned)ROWS << 16);          // tensor_dim0 hi=0 | tensor_dim1 = 32
    g1[3] = (int)((unsigned)(T_ * F_) << 16);     // tensor_dim1 hi=0 | tile_dim0 = 192
    g1[4] = ROWS;                                 // tile_dim1 = 32 | tile_dim2 = 0
    g1[5] = T_ * F_;                              // tensor_dim0_stride = 192
    g1[6] = 0;
    g1[7] = 0;
    i32x4 gz = {0, 0, 0, 0};
#if defined(__clang_major__) && (__clang_major__ >= 23)
    i32x8 g4 = {0, 0, 0, 0, 0, 0, 0, 0};
    __builtin_amdgcn_tensor_load_to_lds(g0, g1, gz, gz, g4, 0);
#else
    __builtin_amdgcn_tensor_load_to_lds(g0, g1, gz, gz, 0);
#endif
}

extern "C" __global__ __launch_bounds__(256)
void lstm_wmma_kernel(const float* __restrict__ seq,    // [N, T, F]
                      const float* __restrict__ Wih,    // [512, 4]
                      const float* __restrict__ Whh,    // [512, 128]
                      const float* __restrict__ bih,    // [512]
                      const float* __restrict__ bhh,    // [512]
                      const float* __restrict__ Wout,   // [4, 128]
                      const float* __restrict__ bo,     // [4]
                      float* __restrict__ out)          // [N, P, F]
{
    extern __shared__ char smem[];
    _Float16* whh  = (_Float16*)(smem + OFF_WHH);
    _Float16* wout = (_Float16*)(smem + OFF_WOUT);
    _Float16* hl   = (_Float16*)(smem + OFF_H);
    float*    seql = (float*)   (smem + OFF_SEQ);
    float*    bias = (float*)   (smem + OFF_BIAS);
    float*    bout = (float*)   (smem + OFF_BOUT);

    const int tid  = threadIdx.x;
    const int wv   = tid >> 5;        // wave 0..7 -> hidden units [16w,16w+16)
    const int lane = tid & 31;
    const int half = lane >> 4;       // hi half -> rows +8 in C layout
    const int ln   = lane & 15;       // column within 16-wide tile
    const int row0 = blockIdx.x * ROWS;

    // ---- async: TDM pulls the whole 24KB sequence tile into LDS ----
    if (wv == 0)
        tdm_load_seq_tile(seq + (size_t)row0 * T_ * F_, (unsigned)OFF_SEQ);

    // ---- stage extended weights [Whh | Wih | 0] as f16 while TDM runs ----
    for (int i = tid; i < G_ * PITCH; i += 256) {
        int j = i / PITCH, k = i % PITCH;
        float v = (k < H_)        ? Whh[j * H_ + k]
                : (k < H_ + F_)   ? Wih[j * F_ + (k - H_)]
                                  : 0.0f;
        whh[i] = (_Float16)v;
    }
    for (int i = tid; i < 16 * PITCH; i += 256) {
        int f = i / PITCH, k = i % PITCH;
        float v = (f < F_ && k < H_) ? Wout[f * H_ + k] : 0.0f;
        wout[i] = (_Float16)v;
    }
    for (int i = tid; i < G_; i += 256) bias[i] = bih[i] + bhh[i];
    if (tid < F_) bout[tid] = bo[tid];
    for (int i = tid; i < ROWS * PITCH; i += 256) hl[i] = (_Float16)0.0f;  // h0=0, x-pad=0

    if (wv == 0) __builtin_amdgcn_s_wait_tensorcnt(0);   // seq tile landed
    __syncthreads();

    // x_0 into extended columns of h
    if (tid < ROWS * F_)
        hl[(tid >> 2) * PITCH + H_ + (tid & 3)] = (_Float16)seql[(tid >> 2) * (T_ * F_) + (tid & 3)];
    __syncthreads();

    // per-wave constants: gate column j and fused bias
    int   jc[4];
    float biasv[4];
#pragma unroll
    for (int ct = 0; ct < 4; ++ct) {
        jc[ct] = ct * H_ + wv * 16 + ln;
        biasv[ct] = bias[jc[ct]];
    }

    float cst[2][8];                                   // c state, C-matrix layout
#pragma unroll
    for (int rt = 0; rt < 2; ++rt)
#pragma unroll
        for (int v = 0; v < 8; ++v) cst[rt][v] = 0.0f;

    // ---------------- one LSTM cell step ----------------
    // gates = [h|x] @ [Whh|Wih]^T + (b_ih+b_hh); xnext (if non-null) is written
    // into the extended x-columns for the NEXT step during the write phase.
    auto cell_step = [&](const float* xnext) {
        v8f acc[2][4];
#pragma unroll
        for (int rt = 0; rt < 2; ++rt)
#pragma unroll
            for (int ct = 0; ct < 4; ++ct)
#pragma unroll
                for (int v = 0; v < 8; ++v) acc[rt][ct][v] = biasv[ct];

        // K = 160: 5 chunks of 32 (f16 WMMA, f32 accumulate)
#pragma unroll
        for (int kc = 0; kc < 5; ++kc) {
            v16h a0 = load_a16(hl + (0  + ln) * PITCH, kc * 32 + half * 16);
            v16h a1 = load_a16(hl + (16 + ln) * PITCH, kc * 32 + half * 16);
#pragma unroll
            for (int ct = 0; ct < 4; ++ct) {
                v16h b = load_b16(whh + jc[ct] * PITCH + kc * 32 + half * 16);
                acc[0][ct] = __builtin_amdgcn_wmma_f32_16x16x32_f16(
                    false, a0, false, b, (short)0, acc[0][ct], false, false);
                acc[1][ct] = __builtin_amdgcn_wmma_f32_16x16x32_f16(
                    false, a1, false, b, (short)0, acc[1][ct], false, false);
            }
        }

        __syncthreads();  // all reads of old [h|x] complete before overwrite

        // torch gate order i,f,g,o ; c' = sig(f)*c + sig(i)*tanh(g) ; h' = sig(o)*tanh(c')
#pragma unroll
        for (int rt = 0; rt < 2; ++rt) {
#pragma unroll
            for (int v = 0; v < 8; ++v) {
                float ig = sigf(acc[rt][0][v]);
                float fg = sigf(acc[rt][1][v]);
                float gg = tanhfast(acc[rt][2][v]);
                float og = sigf(acc[rt][3][v]);
                float c  = __builtin_fmaf(fg, cst[rt][v], ig * gg);
                cst[rt][v] = c;
                float h  = og * tanhfast(c);
                int r = rt * 16 + half * 8 + v;
                hl[r * PITCH + wv * 16 + ln] = (_Float16)h;
            }
        }
        if (xnext && tid < ROWS * F_) {   // stage next step's x (encode path)
            int r = tid >> 2, f = tid & 3;
            hl[r * PITCH + H_ + f] = (_Float16)xnext[r * (T_ * F_) + f];
        }
        __syncthreads();
    };

    // ---------------- encode T steps ----------------
    // step t consumes x_t (already in hl) and stages x_{t+1}; at t=T-1 the
    // x-columns are left holding x_{T-1}, which is exactly decode's first input.
    for (int t = 0; t < T_; ++t)
        cell_step((t + 1 < T_) ? (seql + (t + 1) * F_) : nullptr);

    // ---------------- autoregressive decode P steps ----------------
    for (int p = 0; p < P_; ++p) {
        cell_step(nullptr);

        // out = h @ Wout^T + bo  (waves 0..1, one 16-col WMMA tile, cols 4..15 zero);
        // result goes to global output AND back into the x-columns as next input.
        if (wv < 2) {
            const int rt = wv;
            v8f oa;
#pragma unroll
            for (int v = 0; v < 8; ++v) oa[v] = 0.0f;
#pragma unroll
            for (int kc = 0; kc < 4; ++kc) {     // K=128 only (h part)
                v16h a = load_a16(hl + (rt * 16 + ln) * PITCH, kc * 32 + half * 16);
                v16h b = load_b16(wout + ln * PITCH + kc * 32 + half * 16);
                oa = __builtin_amdgcn_wmma_f32_16x16x32_f16(
                    false, a, false, b, (short)0, oa, false, false);
            }
            if (ln < F_) {
                float bov = bout[ln];
#pragma unroll
                for (int v = 0; v < 8; ++v) {
                    int r = rt * 16 + half * 8 + v;
                    float val = oa[v] + bov;
                    out[(size_t)(row0 + r) * (P_ * F_) + p * F_ + ln] = val;
                    hl[r * PITCH + H_ + ln] = (_Float16)val;   // next step's x
                }
            }
        }
        __syncthreads();
    }
}

extern "C" void kernel_launch(void* const* d_in, const int* in_sizes, int n_in,
                              void* d_out, int out_size, void* d_ws, size_t ws_size,
                              hipStream_t stream) {
    (void)in_sizes; (void)n_in; (void)out_size; (void)d_ws; (void)ws_size;
    // setup_inputs order: 0 sequence, 1 dist, 2 bearing, 3 heading, 4 seq_mask,
    //                     5 op_mask, 6 W_ih, 7 W_hh, 8 b_ih, 9 b_hh, 10 W_out, 11 b_out
    const float* seq  = (const float*)d_in[0];
    const float* Wih  = (const float*)d_in[6];
    const float* Whh  = (const float*)d_in[7];
    const float* bih  = (const float*)d_in[8];
    const float* bhh  = (const float*)d_in[9];
    const float* Wout = (const float*)d_in[10];
    const float* bo   = (const float*)d_in[11];
    float* out = (float*)d_out;

    dim3 grid(N_ / ROWS);   // 256 workgroups, each owns 32 recurrent lanes
    dim3 block(256);        // 8 wave32
    lstm_wmma_kernel<<<grid, block, SMEM_TOTAL, stream>>>(
        seq, Wih, Whh, bih, bhh, Wout, bo, out);
}
